// GCN_88278757802628
// MI455X (gfx1250) — matrix operand order repaired and checked by reading, not compiled, not run on hardware
//
#include <hip/hip_runtime.h>
#include <hip/hip_bf16.h>
#include <stdint.h>

// ---------------- types for WMMA ----------------
typedef __attribute__((ext_vector_type(16))) __bf16 v16bf;
typedef __attribute__((ext_vector_type(8)))  float  v8f;

__device__ __forceinline__ uint16_t f2bf(float f) {
    uint32_t u = __builtin_bit_cast(uint32_t, f);
    u += 0x7FFFu + ((u >> 16) & 1u);          // round-to-nearest-even
    return (uint16_t)(u >> 16);
}
__device__ __forceinline__ uint32_t pack2bf(float lo, float hi) {
    return (uint32_t)f2bf(lo) | ((uint32_t)f2bf(hi) << 16);
}

// gfx1250 async copy: global -> LDS, 16B per lane, tracked by ASYNCcnt.
__device__ __forceinline__ void async_cp_b128(uint32_t lds_off, const void* gaddr) {
    asm volatile("global_load_async_to_lds_b128 %0, %1, off"
                 :: "v"(lds_off), "v"(gaddr) : "memory");
}

#define TM 128
#define TN 64
#define TK 32
// LDS tile: A 128 rows x 64B (8KB) + B 64 rows x 64B (4KB) = 12KB, double buffered.
#define ATILE_B   8192
#define TILE_B    12288

// C(f32, MxN) = A(bf16, MxK row-major) * Wt(bf16, NxK row-major, i.e. W transposed)
// 256 threads (8 waves), block tile 128x64, wave tile 16x64 (4 accum frags).
// Tiles staged with GLOBAL_LOAD_ASYNC_TO_LDS_B128, double-buffered, XOR-swizzled.
__global__ __launch_bounds__(256) void gemm_bf16_wmma_async(
    const uint16_t* __restrict__ A, const uint16_t* __restrict__ Wt,
    float* __restrict__ Out, int M, int N, int K)
{
    __shared__ alignas(16) uint8_t smem[2 * TILE_B];
    const uint32_t smem32 = (uint32_t)(uintptr_t)smem;   // LDS byte offset (low 32 bits)

    const int tid  = threadIdx.x;
    const int wave = tid >> 5;
    const int lane = tid & 31;
    const int hlf  = lane >> 4;
    const int l15  = lane & 15;
    const int m0 = blockIdx.x * TM;
    const int n0 = blockIdx.y * TN;

    // ---- staging map: one 16B block per slot, swizzle blk ^= (row>>2)&3 ----
    const int rA  = tid >> 2;            // rows rA and rA+64 of the A tile
    const int bA  = tid & 3;
    const int swA = (rA >> 2) & 3;       // same swizzle for rA+64
    const uint32_t offA0 = (uint32_t)( rA       * 64 + ((bA ^ swA) << 4));
    const uint32_t offA1 = (uint32_t)((rA + 64) * 64 + ((bA ^ swA) << 4));
    const uint32_t offB  = (uint32_t)(ATILE_B + rA * 64 + ((bA ^ swA) << 4));
    const bool okA0 = (m0 + rA)      < M;
    const bool okA1 = (m0 + rA + 64) < M;
    // OOB rows load a safe dummy row (row 0): their C rows are discarded at store,
    // and the per-wave async issue count stays uniform (3 per stage).
    const uint8_t* gA0 = (const uint8_t*)(A + (okA0 ? (size_t)(m0 + rA)      * K : 0) + bA * 8);
    const uint8_t* gA1 = (const uint8_t*)(A + (okA1 ? (size_t)(m0 + rA + 64) * K : 0) + bA * 8);
    const uint8_t* gB  = (const uint8_t*)(Wt + (size_t)(n0 + rA) * K + bA * 8);

    // ---- fragment LDS offsets (constant across the K loop) ----
    // A frag (ISA 7.12.2, 16-bit A 16x32): lane<16 K base 0, lane>=16 K base 8.
    // u[0..3] = K {0,2,4,6}+8*hlf (one 16B block), u[4..7] = +16 (next block).
    const int arow = wave * 16 + l15;
    const int aswz = (arow >> 2) & 3;
    const uint32_t fA0 = (uint32_t)(arow * 64 + (((hlf    ) ^ aswz) << 4));
    const uint32_t fA1 = (uint32_t)(arow * 64 + (((hlf + 2) ^ aswz) << 4));
    // B frag: same K order as A: blocks hlf and hlf+2 (K 8*hlf..+7, 16+8*hlf..+7).
    uint32_t fB0[4], fB1[4];
    #pragma unroll
    for (int f = 0; f < 4; ++f) {
        const int brow = f * 16 + l15;
        const int bswz = (brow >> 2) & 3;
        fB0[f] = (uint32_t)(ATILE_B + brow * 64 + (((hlf    ) ^ bswz) << 4));
        fB1[f] = (uint32_t)(ATILE_B + brow * 64 + (((hlf + 2) ^ bswz) << 4));
    }

    v8f acc[4];
    #pragma unroll
    for (int f = 0; f < 4; ++f) acc[f] = (v8f){0.f,0.f,0.f,0.f,0.f,0.f,0.f,0.f};

    const int nk = K / TK;
    auto stage = [&](int kt, int buf) {
        const uint32_t bofs = (uint32_t)buf * TILE_B;
        const size_t   go   = (size_t)kt * (TK * 2);   // 64B per K-step
        async_cp_b128(smem32 + bofs + offA0, gA0 + go);
        async_cp_b128(smem32 + bofs + offA1, gA1 + go);
        async_cp_b128(smem32 + bofs + offB,  gB  + go);
    };

    stage(0, 0);
    for (int kt = 0; kt < nk; ++kt) {
        const int cur = kt & 1;
        if (kt + 1 < nk) {
            stage(kt + 1, cur ^ 1);                       // overlap next tile's DMA
            asm volatile("s_wait_asynccnt 0x3" ::: "memory");  // tile kt done (in-order)
        } else {
            asm volatile("s_wait_asynccnt 0x0" ::: "memory");
        }
        __syncthreads();

        const uint8_t* sp = smem + cur * TILE_B;
        union { uint4 q[2]; v16bf v; } af;
        af.q[0] = *(const uint4*)(sp + fA0);
        af.q[1] = *(const uint4*)(sp + fA1);
        #pragma unroll
        for (int f = 0; f < 4; ++f) {
            union { uint4 q[2]; v16bf v; } bfr;
            bfr.q[0] = *(const uint4*)(sp + fB0[f]);
            bfr.q[1] = *(const uint4*)(sp + fB1[f]);
            acc[f] = __builtin_amdgcn_wmma_f32_16x16x32_bf16(
                false, af.v, false, bfr.v, (short)0, acc[f], false, false);
        }
        __syncthreads();
    }

    // ---- store C (vgpr e -> M=e lanes 0-15 / M=8+e lanes 16-31; N=lane&15) ----
    const int mbase = m0 + wave * 16 + 8 * hlf;
    #pragma unroll
    for (int f = 0; f < 4; ++f) {
        const int col = n0 + f * 16 + l15;
        #pragma unroll
        for (int e = 0; e < 8; ++e) {
            int row = mbase + e;
            if (row < M) Out[(size_t)row * N + col] = acc[f][e];
        }
    }
}

// agg[dst[e]] += m[src[e]]; float4 per thread, HW f32 atomics (L2-resident buffers).
__global__ __launch_bounds__(256) void scatter_edges(
    const float* __restrict__ m, const int* __restrict__ src,
    const int* __restrict__ dst, float* __restrict__ agg, int E, int D)
{
    const int D4  = D >> 2;
    int idx = blockIdx.x * blockDim.x + threadIdx.x;
    int e   = idx / D4;
    if (e >= E) return;
    int c = (idx - e * D4) * 4;
    int s = src[e], d = dst[e];
    float4 v = *(const float4*)&m[(size_t)s * D + c];
    float* p = &agg[(size_t)d * D + c];
    unsafeAtomicAdd(p + 0, v.x);
    unsafeAtomicAdd(p + 1, v.y);
    unsafeAtomicAdd(p + 2, v.z);
    unsafeAtomicAdd(p + 3, v.w);
}

// h_bf16 = bf16(relu(agg + b)) : fused epilogue + next-layer operand conversion
__global__ __launch_bounds__(256) void bias_relu_to_bf16(
    const float* __restrict__ agg, const float* __restrict__ b,
    uint16_t* __restrict__ out, int total2, int D)
{
    int i = blockIdx.x * blockDim.x + threadIdx.x;   // pair index
    if (i >= total2) return;
    int j = i * 2;
    float v0 = agg[j]     + b[j       & (D - 1)];
    float v1 = agg[j + 1] + b[(j + 1) & (D - 1)];
    v0 = v0 < 0.f ? 0.f : v0;
    v1 = v1 < 0.f ? 0.f : v1;
    ((uint32_t*)out)[i] = pack2bf(v0, v1);
}

// z = agg + b (f32 in-place, last layer: no relu, decoder wants f32)
__global__ __launch_bounds__(256) void bias_f32(
    float* __restrict__ buf, const float* __restrict__ b, int total, int D)
{
    int i = blockIdx.x * blockDim.x + threadIdx.x;
    if (i >= total) return;
    buf[i] = buf[i] + b[i & (D - 1)];
}

// x -> bf16 (4 elems / thread)
__global__ __launch_bounds__(256) void f32_to_bf16(
    const float* __restrict__ src, uint16_t* __restrict__ dst, int total4)
{
    int i = blockIdx.x * blockDim.x + threadIdx.x;
    if (i >= total4) return;
    float4 v = ((const float4*)src)[i];
    ((uint2*)dst)[i] = make_uint2(pack2bf(v.x, v.y), pack2bf(v.z, v.w));
}

// Wt[n][k] = bf16(W[k][n]) ; one k-pair per thread (packed u32 store)
__global__ __launch_bounds__(256) void w_transpose_bf16(
    const float* __restrict__ W, uint16_t* __restrict__ Wt, int K, int N, int total)
{
    int i = blockIdx.x * blockDim.x + threadIdx.x;   // over N * K/2
    if (i >= total) return;
    int k2n = K >> 1;
    int n  = i / k2n;
    int k2 = i - n * k2n;
    ((uint32_t*)Wt)[(size_t)n * k2n + k2] =
        pack2bf(W[(size_t)(2 * k2)     * N + n],
                W[(size_t)(2 * k2 + 1) * N + n]);
}

// one wave32 per labeled edge: dot(z[s], z[d]) over D floats
__global__ __launch_bounds__(256) void decode_dot(
    const float* __restrict__ z, const int* __restrict__ ls,
    const int* __restrict__ ld, float* __restrict__ out, int L, int D)
{
    int wid  = (blockIdx.x * blockDim.x + threadIdx.x) >> 5;
    int lane = threadIdx.x & 31;
    if (wid >= L) return;
    const float4* za = (const float4*)&z[(size_t)ls[wid] * D];
    const float4* zb = (const float4*)&z[(size_t)ld[wid] * D];
    const int D4 = D >> 2;
    float s = 0.f;
    for (int c = lane; c < D4; c += 32) {
        float4 a = za[c], b = zb[c];
        s += a.x*b.x + a.y*b.y + a.z*b.z + a.w*b.w;
    }
    #pragma unroll
    for (int off = 16; off > 0; off >>= 1) s += __shfl_down(s, off, 32);
    if (lane == 0) out[wid] = s;
}

extern "C" void kernel_launch(void* const* d_in, const int* in_sizes, int n_in,
                              void* d_out, int out_size, void* d_ws, size_t ws_size,
                              hipStream_t stream)
{
    const float* x  = (const float*)d_in[0];
    const int* ei   = (const int*)d_in[1];
    const int* eli  = (const int*)d_in[2];
    const float* W1 = (const float*)d_in[3];
    const float* b1 = (const float*)d_in[4];
    const float* W2 = (const float*)d_in[5];
    const float* b2 = (const float*)d_in[6];
    const float* W3 = (const float*)d_in[7];
    const float* b3 = (const float*)d_in[8];
    float* out = (float*)d_out;

    const int DH = in_sizes[4];          // 512
    const int DI = in_sizes[3] / DH;     // 512
    const int DO = in_sizes[8];          // 256
    const int M  = in_sizes[0] / DI;     // 20000
    const int E  = in_sizes[1] / 2;      // 320000
    const int L  = in_sizes[2] / 2;      // 100000
    const int* src = ei;
    const int* dst = ei + E;
    const int* ls  = eli;
    const int* ld  = eli + L;

    // ---- workspace carve-up (bytes) ----
    uint8_t* ws = (uint8_t*)d_ws;
    float*    bufA = (float*)ws;    ws += (size_t)M * DH * 4;   // messages
    float*    bufB = (float*)ws;    ws += (size_t)M * DH * 4;   // aggregate
    uint16_t* hbf  = (uint16_t*)ws; ws += (size_t)M * DH * 2;   // bf16 A-operand
    uint16_t* wt1  = (uint16_t*)ws; ws += (size_t)DI * DH * 2;  // bf16 W^T
    uint16_t* wt2  = (uint16_t*)ws; ws += (size_t)DH * DH * 2;
    uint16_t* wt3  = (uint16_t*)ws;

    const dim3 blk(256);
    const dim3 g1((M + TM - 1) / TM, DH / TN);
    const dim3 g3((M + TM - 1) / TM, DO / TN);
    const int scat12 = (E * (DH / 4) + 255) / 256;
    const int scat3  = (E * (DO / 4) + 255) / 256;
    const int ep12   = (M * DH / 2 + 255) / 256;

    // ---- one-time per call: operand conversion ----
    f32_to_bf16<<<(M * DI / 4 + 255) / 256, blk, 0, stream>>>(x, hbf, M * DI / 4);
    w_transpose_bf16<<<(DH * DI / 2 + 255) / 256, blk, 0, stream>>>(W1, wt1, DI, DH, DH * DI / 2);
    w_transpose_bf16<<<(DH * DH / 2 + 255) / 256, blk, 0, stream>>>(W2, wt2, DH, DH, DH * DH / 2);
    w_transpose_bf16<<<(DO * DH / 2 + 255) / 256, blk, 0, stream>>>(W3, wt3, DH, DO, DO * DH / 2);

    // ---- layer 1 ----
    gemm_bf16_wmma_async<<<g1, blk, 0, stream>>>(hbf, wt1, bufA, M, DH, DI);
    hipMemsetAsync(bufB, 0, (size_t)M * DH * sizeof(float), stream);
    scatter_edges<<<scat12, blk, 0, stream>>>(bufA, src, dst, bufB, E, DH);
    bias_relu_to_bf16<<<ep12, blk, 0, stream>>>(bufB, b1, hbf, M * DH / 2, DH);

    // ---- layer 2 ----
    gemm_bf16_wmma_async<<<g1, blk, 0, stream>>>(hbf, wt2, bufA, M, DH, DH);
    hipMemsetAsync(bufB, 0, (size_t)M * DH * sizeof(float), stream);
    scatter_edges<<<scat12, blk, 0, stream>>>(bufA, src, dst, bufB, E, DH);
    bias_relu_to_bf16<<<ep12, blk, 0, stream>>>(bufB, b2, hbf, M * DH / 2, DH);

    // ---- layer 3 ----
    gemm_bf16_wmma_async<<<g3, blk, 0, stream>>>(hbf, wt3, bufA, M, DO, DH);
    hipMemsetAsync(bufB, 0, (size_t)M * DO * sizeof(float), stream);
    scatter_edges<<<scat3, blk, 0, stream>>>(bufA, src, dst, bufB, E, DO);
    bias_f32<<<(M * DO + 255) / 256, blk, 0, stream>>>(bufB, b3, M * DO, DO);

    // ---- decode ----
    decode_dot<<<(L * 32 + 255) / 256, blk, 0, stream>>>(bufB, ls, ld, out, L, DO);
}
